// FlowNet_38543036514796
// MI455X (gfx1250) — matrix operand order repaired and checked by reading, not compiled, not run on hardware
//
#include <hip/hip_runtime.h>
#include <math.h>

typedef __attribute__((ext_vector_type(16))) _Float16 v16h;
typedef __attribute__((ext_vector_type(2)))  _Float16 v2h;
typedef __attribute__((ext_vector_type(8)))  float    v8f;

#define LRELU(v, s) (((v) >= 0.f) ? (v) : (s) * (v))

// ---------------------------------------------------------------------------
// Implicit-GEMM 3x3 conv (pad=1) via WMMA f16 -> f32, with virtual 3-source
// channel concatenation folded into the im2col gather.
// Block = 256 threads = 8 wave32s. Tile = 32 out-channels x 128 pixels.
// Each wave computes two 16x16 WMMA tiles (shared A fragment); K = Cin*9
// reduced in chunks of 32. LDS holds operands in *fragment order* so each
// lane reads its v16h operand with two ds_load_b128; stagers perform the
// CDNA5 A/B VGPR-striping shuffle at write time with packed ds_store_b32.
// Staging is split into a pure-load phase (all global loads in flight) and a
// convert/store phase; out-of-bounds handled by clamped addresses * 0-mask
// (no exec divergence). Epilogue fuses bias + leaky-ReLU + residual add.
// ---------------------------------------------------------------------------
__global__ __launch_bounds__(256, 2)
void conv3x3_wmma_kernel(const float* __restrict__ X0, const float* __restrict__ X1,
                         const float* __restrict__ X2, int C0, int C01,
                         const float* __restrict__ Wt,
                         const float* __restrict__ Bias, const float* __restrict__ Res,
                         float* __restrict__ Y,
                         int N, int Cin, int Cout, int Hin, int Win,
                         int Hout, int Wout, int stride, float slope, int relu)
{
    const int K   = Cin * 9;
    const int HWo = Hout * Wout;
    const int P   = N * HWo;

    // fragment-order LDS: [fragment][lane][element]
    __shared__ __align__(32) _Float16 sA[2][32][16];   // 2 A fragments (waveM)   = 2 KB
    __shared__ __align__(32) _Float16 sB[8][32][16];   // 8 B fragments (px tile) = 8 KB

    const int tid   = threadIdx.x;
    const int lane  = tid & 31;
    const int wave  = tid >> 5;
    const int waveM = wave & 1;       // which 16-channel half of the 32-ch tile
    const int waveN = wave >> 1;      // which pair of 16-pixel tiles
    const int lr    = lane & 15;

    const int pix0  = blockIdx.x * 128;
    const int cout0 = blockIdx.y * 32;

    v8f acc0 = {}, acc1 = {};

    for (int k0 = 0; k0 < K; k0 += 32) {
        // ================= phase 1: issue ALL global loads =================
        float av[4];
        #pragma unroll
        for (int it = 0; it < 2; ++it) {
            int i   = tid + it * 256;          // pair index (512 pairs)
            int co  = i >> 4;
            int kk  = (i & 15) << 1;           // even k within chunk
            int gco = cout0 + co;
            int gk  = k0 + kk;
            int cco = (gco < Cout) ? gco : (Cout - 1);
            const float* wp = Wt + (size_t)cco * K;    // OIHW flat: k = ci*9 + r*3 + s
            float m  = (gco < Cout) ? 1.f : 0.f;
            float m0 = (gk     < K) ? m : 0.f;
            float m1 = (gk + 1 < K) ? m : 0.f;
            av[2 * it + 0] = wp[(gk     < K) ? gk     : (K - 1)] * m0;  // adjacent pair
            av[2 * it + 1] = wp[(gk + 1 < K) ? gk + 1 : (K - 1)] * m1;
        }
        float bv[16];
        #pragma unroll
        for (int it = 0; it < 8; ++it) {
            int i   = tid + it * 256;          // pair index (2048 pairs)
            int p   = i & 127;
            int kk  = (i >> 7) << 1;           // even k within chunk
            int gp  = pix0 + p;
            int cgp = (gp < P) ? gp : (P - 1);
            int n   = cgp / HWo;
            int rem = cgp - n * HWo;
            int oy  = rem / Wout, ox = rem - oy * Wout;
            #pragma unroll
            for (int q = 0; q < 2; ++q) {
                int gk  = k0 + kk + q;
                int cgk = (gk < K) ? gk : (K - 1);
                int ci  = cgk / 9, t = cgk - ci * 9;
                int r   = t / 3,   s = t - r * 3;
                int iy  = oy * stride + r - 1;
                int ix  = ox * stride + s - 1;
                float m = ((gp < P) & (gk < K) & (iy >= 0) & (iy < Hin) &
                           (ix >= 0) & (ix < Win)) ? 1.f : 0.f;
                int cy = iy < 0 ? 0 : (iy >= Hin ? Hin - 1 : iy);
                int cx = ix < 0 ? 0 : (ix >= Win ? Win - 1 : ix);
                // virtual concat: pick source tensor / local channel per K index
                const float* Xs = (ci < C0) ? X0 : ((ci < C01) ? X1 : X2);
                int cc = (ci < C0) ? ci : ((ci < C01) ? (ci - C0) : (ci - C01));
                int Cs = (ci < C0) ? C0 : ((ci < C01) ? (C01 - C0) : (Cin - C01));
                bv[2 * it + q] =
                    Xs[(((size_t)n * Cs + cc) * Hin + cy) * Win + cx] * m;  // unconditional
            }
        }
        // CDNA5 prefetch of next weight chunk (global_prefetch_b8)
        if (k0 + 32 < K) {
            int pco = cout0 + (tid >> 3); if (pco > Cout - 1) pco = Cout - 1;
            int pk  = k0 + 32 + ((tid & 7) << 2); if (pk > K - 1) pk = K - 1;
            __builtin_prefetch(&Wt[(size_t)pco * K + pk], 0, 1);
        }

        // ================= phase 2: convert + fragment-order LDS stores ====
        // A-fragment layout: lanes 0-15 K 0..7 / 16..23, lanes 16-31 K 8..15 / 24..31
        #pragma unroll
        for (int it = 0; it < 2; ++it) {
            int i  = tid + it * 256;
            int co = i >> 4;
            int kk = (i & 15) << 1;
            int h  = (kk >> 3) & 1;
            int e  = (kk & 7) + ((kk >> 4) << 3);
            v2h pkt = { (_Float16)av[2 * it + 0], (_Float16)av[2 * it + 1] };
            *(v2h*)&sA[co >> 4][h * 16 + (co & 15)][e] = pkt;
        }
        // B-fragment layout: lanes 0-15 K 0..7 / 16..23, lanes 16-31 K 8..15 / 24..31
        #pragma unroll
        for (int it = 0; it < 8; ++it) {
            int i  = tid + it * 256;
            int p  = i & 127;
            int kk = (i >> 7) << 1;
            v2h pkt = { (_Float16)bv[2 * it + 0], (_Float16)bv[2 * it + 1] };
            *(v2h*)&sB[p >> 4][(((kk >> 3) & 1) << 4) + (p & 15)][(kk & 7) + ((kk >> 4) << 3)] = pkt;
        }
        __syncthreads();

        // ---- contiguous fragment loads (2x ds_load_b128 each) + 2 WMMAs ----
        v16h a  = *(const v16h*)(&sA[waveM][lane][0]);
        v16h b0 = *(const v16h*)(&sB[waveN * 2 + 0][lane][0]);
        v16h b1 = *(const v16h*)(&sB[waveN * 2 + 1][lane][0]);
        acc0 = __builtin_amdgcn_wmma_f32_16x16x32_f16(false, a, false, b0, (short)0, acc0, false, false);
        acc1 = __builtin_amdgcn_wmma_f32_16x16x32_f16(false, a, false, b1, (short)0, acc1, false, false);
        __syncthreads();
    }

    // ---- epilogue: D layout M = r + 8*(lane/16), N = lane%16 ----
    const int half = lane >> 4;
    #pragma unroll
    for (int j = 0; j < 2; ++j) {
        v8f acc = j ? acc1 : acc0;
        int gp = pix0 + (waveN * 2 + j) * 16 + lr;
        if (gp < P) {
            int n   = gp / HWo;
            int rem = gp - n * HWo;
            int oy  = rem / Wout, ox = rem - oy * Wout;
            #pragma unroll
            for (int r = 0; r < 8; ++r) {
                int co = cout0 + waveM * 16 + r + 8 * half;
                if (co < Cout) {
                    float v = acc[r] + Bias[co];
                    size_t oidx = (((size_t)n * Cout + co) * Hout + oy) * Wout + ox;
                    if (Res) v += Res[oidx];
                    if (relu) v = LRELU(v, slope);
                    Y[oidx] = v;
                }
            }
        }
    }
}

// ---------------------------------------------------------------------------
// 7x7 correlation volume (49 channels), mean over C, fused lrelu(0.01)
// ---------------------------------------------------------------------------
__global__ void correlation_kernel(const float* __restrict__ A, const float* __restrict__ Bf,
                                   float* __restrict__ Out, int N, int C, int H, int W)
{
    int idx = blockIdx.x * blockDim.x + threadIdx.x;
    int total = N * 49 * H * W;
    if (idx >= total) return;
    int x = idx % W; int t = idx / W;
    int y = t % H;   t /= H;
    int d = t % 49;  int n = t / 49;
    int dy = d / 7 - 3, dx = d % 7 - 3;
    int yy = y + dy, xx = x + dx;
    float s = 0.f;
    if (yy >= 0 && yy < H && xx >= 0 && xx < W) {
        const float* ap = A  + ((size_t)n * C * H + y)  * W + x;
        const float* bp = Bf + ((size_t)n * C * H + yy) * W + xx;
        size_t cs = (size_t)H * W;
        for (int c = 0; c < C; ++c) s += ap[c * cs] * bp[c * cs];
    }
    s /= (float)C;
    Out[idx] = LRELU(s, 0.01f);
}

// ---------------------------------------------------------------------------
// Backward warp: reflect-pad bilinear sampling driven by flow (reference "project")
// ---------------------------------------------------------------------------
__global__ void project_kernel(const float* __restrict__ F, const float* __restrict__ U,
                               float* __restrict__ Out, int N, int C, int H, int W, float dt)
{
    int idx = blockIdx.x * blockDim.x + threadIdx.x;
    int total = N * C * H * W;
    if (idx >= total) return;
    int x = idx % W; int t = idx / W;
    int y = t % H;   t /= H;
    int c = t % C;   int n = t / C;

    float ux = U[(((size_t)n * 2 + 1) * H + y) * W + x] * (2.0f / (H - 1.0f));
    float uy = U[(((size_t)n * 2 + 0) * H + y) * W + x] * (2.0f / (W - 1.0f));
    float gx = -1.0f + 2.0f * (float)x / (W - 1.0f);
    float gy = -1.0f + 2.0f * (float)y / (H - 1.0f);
    float sx = gx - ux * dt;
    float sy = gy - uy * dt;
    float fx = (sx + 1.0f) * 0.5f * (W - 1.0f);
    float fy = (sy + 1.0f) * 0.5f * (H - 1.0f);
    float spx = W - 1.0f, spy = H - 1.0f;
    float mx = fmodf(fabsf(fx), 2.0f * spx); mx = (mx > spx) ? (2.0f * spx - mx) : mx;
    float my = fmodf(fabsf(fy), 2.0f * spy); my = (my > spy) ? (2.0f * spy - my) : my;
    int x0 = (int)floorf(mx), y0 = (int)floorf(my);
    float wx = mx - (float)x0, wy = my - (float)y0;
    x0 = max(0, min(x0, W - 1)); int x1 = min(x0 + 1, W - 1);
    y0 = max(0, min(y0, H - 1)); int y1 = min(y0 + 1, H - 1);
    const float* img = F + ((size_t)n * C + c) * H * W;
    float v00 = img[y0 * W + x0], v01 = img[y0 * W + x1];
    float v10 = img[y1 * W + x0], v11 = img[y1 * W + x1];
    Out[idx] = (1.f - wy) * ((1.f - wx) * v00 + wx * v01) + wy * ((1.f - wx) * v10 + wx * v11);
}

// ---------------------------------------------------------------------------
// Grouped 4x4 stride-2 transposed conv (flow 2ch, groups=2, kernel flipped)
// ---------------------------------------------------------------------------
__global__ void flow_upsample_kernel(const float* __restrict__ Fl, const float* __restrict__ Wup,
                                     float* __restrict__ Out, int N, int H, int W)
{
    int Ho = 2 * H, Wo = 2 * W;
    int idx = blockIdx.x * blockDim.x + threadIdx.x;
    int total = N * 2 * Ho * Wo;
    if (idx >= total) return;
    int ox = idx % Wo; int t = idx / Wo;
    int oy = t % Ho;   t /= Ho;
    int g  = t % 2;    int n = t / 2;
    float s = 0.f;
    for (int r = 0; r < 4; ++r) {
        int ty = oy - 2 + r;
        if (ty < 0 || ty > 2 * H - 2 || (ty & 1)) continue;
        int iy = ty >> 1;
        for (int q = 0; q < 4; ++q) {
            int tx = ox - 2 + q;
            if (tx < 0 || tx > 2 * W - 2 || (tx & 1)) continue;
            int ix = tx >> 1;
            float wv = Wup[((size_t)g * 16) + (3 - r) * 4 + (3 - q)];  // flipped kernel
            s += wv * Fl[(((size_t)n * 2 + g) * H + iy) * W + ix];
        }
    }
    Out[idx] = s;
}

// ---------------------------------------------------------------------------
// Bilinear resize (half-pixel centers, matches jax.image.resize 'linear')
// ---------------------------------------------------------------------------
__global__ void resize_bilinear_kernel(const float* __restrict__ In, float* __restrict__ Out,
                                       int N, int C, int Hi, int Wi, int Ho, int Wo)
{
    int idx = blockIdx.x * blockDim.x + threadIdx.x;
    int total = N * C * Ho * Wo;
    if (idx >= total) return;
    int x = idx % Wo; int t = idx / Wo;
    int y = t % Ho;   t /= Ho;
    int c = t % C;    int n = t / C;
    float sy = ((float)y + 0.5f) * (float)Hi / (float)Ho - 0.5f;
    float sx = ((float)x + 0.5f) * (float)Wi / (float)Wo - 0.5f;
    sy = fmaxf(0.f, fminf(sy, Hi - 1.0f));
    sx = fmaxf(0.f, fminf(sx, Wi - 1.0f));
    int y0 = (int)floorf(sy), x0 = (int)floorf(sx);
    int y1 = min(y0 + 1, Hi - 1), x1 = min(x0 + 1, Wi - 1);
    float wy = sy - y0, wx = sx - x0;
    const float* img = In + ((size_t)n * C + c) * Hi * Wi;
    float v00 = img[y0 * Wi + x0], v01 = img[y0 * Wi + x1];
    float v10 = img[y1 * Wi + x0], v11 = img[y1 * Wi + x1];
    Out[idx] = (1.f - wy) * ((1.f - wx) * v00 + wx * v01) + wy * ((1.f - wx) * v10 + wx * v11);
}

// ===========================================================================
// Host orchestration
// ===========================================================================
struct CP { const float* w; const float* b; };

// up-to-3-source conv launch (virtual concat along channels)
static inline void launch_conv3(const float* X0, const float* X1, const float* X2,
                                int C0, int C1, const float* Wt, const float* Bi,
                                const float* Res, float* Y,
                                int N, int Cin, int Cout, int Hin, int Win,
                                int Hout, int Wout, int stride, float slope, int relu,
                                hipStream_t st)
{
    int P = N * Hout * Wout;
    dim3 grid((P + 127) / 128, (Cout + 31) / 32);
    conv3x3_wmma_kernel<<<grid, dim3(256), 0, st>>>(X0, X1 ? X1 : X0, X2 ? X2 : X0,
                                                    C0, C0 + C1, Wt, Bi, Res, Y,
                                                    N, Cin, Cout, Hin, Win,
                                                    Hout, Wout, stride, slope, relu);
}

static inline void launch_conv(const float* X, const float* Wt, const float* Bi,
                               const float* Res, float* Y,
                               int N, int Cin, int Cout, int Hin, int Win,
                               int Hout, int Wout, int stride, float slope, int relu,
                               hipStream_t st)
{
    launch_conv3(X, nullptr, nullptr, Cin, 0, Wt, Bi, Res, Y,
                 N, Cin, Cout, Hin, Win, Hout, Wout, stride, slope, relu, st);
}

static inline dim3 g1(int total) { return dim3((total + 255) / 256); }

// conv chain; first layer may virtually concat 3 sources
static void field_conv_host(const float* s0, const float* s1, const float* s2,
                            int c0, int c1, const int* chans, int nl, const CP* ps,
                            const float* residual, float* outp,
                            float* tA, float* tB, int B, int H, hipStream_t st)
{
    for (int l = 0; l < nl; ++l) {
        bool last = (l == nl - 1);
        float* dst = last ? outp : ((l & 1) ? tB : tA);
        if (l == 0)
            launch_conv3(s0, s1, s2, c0, c1, ps[l].w, ps[l].b, last ? residual : nullptr, dst,
                         B, chans[0], chans[1], H, H, H, H, 1, 0.1f, last ? 0 : 1, st);
        else
            launch_conv((l & 1) ? tA : tB, ps[l].w, ps[l].b, last ? residual : nullptr, dst,
                        B, chans[l], chans[l + 1], H, H, H, H, 1, 0.1f, last ? 0 : 1, st);
    }
}

extern "C" void kernel_launch(void* const* d_in, const int* in_sizes, int n_in,
                              void* d_out, int out_size, void* d_ws, size_t ws_size,
                              hipStream_t stream)
{
    const int B = 8;
    const int FEAT[4] = {32, 64, 96, 128};

    const float* f1 = (const float*)d_in[0];
    const float* f2 = (const float*)d_in[1];
    // d_in[2] (coord) and d_in[3] (t) are unused by reference()

    int pi = 4;
    auto nxt = [&]() -> const float* {
        int i = (pi < n_in) ? pi : (n_in - 1);
        ++pi;
        return (const float*)d_in[i];
    };

    const float *fw1[4], *fb1[4], *fw2[4], *fb2[4];
    for (int j = 0; j < 4; ++j) { fw1[j] = nxt(); fb1[j] = nxt(); fw2[j] = nxt(); fb2[j] = nxt(); }
    const float* upw[4];
    CP corrp[4][4], refp[4][4];
    for (int i = 0; i < 4; ++i) {
        upw[i] = nxt();
        for (int l = 0; l < 4; ++l) { corrp[i][l].w = nxt(); corrp[i][l].b = nxt(); }
        for (int l = 0; l < 4; ++l) { refp[i][l].w  = nxt(); refp[i][l].b  = nxt(); }
    }
    CP upp[3];
    for (int l = 0; l < 3; ++l) { upp[l].w = nxt(); upp[l].b = nxt(); }

    // ---- workspace carving ----
    char* base = (char*)d_ws; size_t off = 0;
    auto carve = [&](size_t elems) -> float* {
        float* p = (float*)(base + off);
        off += ((elems * sizeof(float) + 255) & ~(size_t)255);
        return p;
    };
    float *feats1[4], *feats2[4];
    for (int j = 0; j < 4; ++j) { int H = 128 >> j; feats1[j] = carve((size_t)B * FEAT[j] * H * H); }
    for (int j = 0; j < 4; ++j) { int H = 128 >> j; feats2[j] = carve((size_t)B * FEAT[j] * H * H); }
    float* tmpA  = carve((size_t)B * 64 * 256 * 256);   // largest intermediate (64ch@256², 128ch@128²)
    float* tmpB  = carve((size_t)B * 32 * 256 * 256);
    float* corrb = carve((size_t)B * 49 * 128 * 128);
    float* warp  = carve((size_t)B * 32 * 128 * 128);   // max C*H² across levels
    float* flow1 = carve((size_t)B * 2 * 128 * 128);
    float* flowu = carve((size_t)B * 2 * 128 * 128);
    float* xres  = carve((size_t)B * 2 * 256 * 256);

    float* out = (float*)d_out;
    size_t oofs[5];
    oofs[0] = 0;
    oofs[1] = oofs[0] + (size_t)B * 2 * 16 * 16;
    oofs[2] = oofs[1] + (size_t)B * 2 * 32 * 32;
    oofs[3] = oofs[2] + (size_t)B * 2 * 64 * 64;
    oofs[4] = oofs[3] + (size_t)B * 2 * 128 * 128;

    // ---- feature pyramid for both images ----
    for (int img = 0; img < 2; ++img) {
        const float* cur = img ? f2 : f1;
        float** feats = img ? feats2 : feats1;
        int Cin = 1, Hin = 256;
        for (int j = 0; j < 4; ++j) {
            int Hout = Hin >> 1;
            launch_conv(cur, fw1[j], fb1[j], nullptr, tmpA, B, Cin, FEAT[j],
                        Hin, Hin, Hout, Hout, 2, 0.1f, 1, stream);
            launch_conv(tmpA, fw2[j], fb2[j], nullptr, feats[j], B, FEAT[j], FEAT[j],
                        Hout, Hout, Hout, Hout, 1, 0.1f, 1, stream);
            cur = feats[j]; Cin = FEAT[j]; Hin = Hout;
        }
    }

    // ---- cascaded flow estimation (level 3 -> 0) ----
    const int corr_ch[5] = {49, 128, 64, 32, 2};
    for (int i = 0; i < 4; ++i) {
        int level = 3 - i;
        int H = 16 << i;
        int C = FEAT[level];
        float dt_m = ldexpf(1.0f, -level);   // DT * 0.5^level
        float dt_s = 0.5f * dt_m;
        float* fout = out + oofs[i];
        const float* fe1 = feats1[level];
        const float* fe2 = feats2[level];

        const float* base_flow = nullptr;
        const float* corr_b = fe2;
        if (i > 0) {
            int Hp = H >> 1;
            flow_upsample_kernel<<<g1(B * 2 * H * H), 256, 0, stream>>>(
                out + oofs[i - 1], upw[i], flowu, B, Hp, Hp);
            project_kernel<<<g1(B * C * H * H), 256, 0, stream>>>(
                fe2, flowu, warp, B, C, H, H, -dt_m);
            corr_b = warp; base_flow = flowu;
        }
        correlation_kernel<<<g1(B * 49 * H * H), 256, 0, stream>>>(fe1, corr_b, corrb, B, C, H, H);
        field_conv_host(corrb, nullptr, nullptr, 49, 0, corr_ch, 4, corrp[i],
                        base_flow, flow1, tmpA, tmpB, B, H, stream);

        project_kernel<<<g1(B * C * H * H), 256, 0, stream>>>(fe2, flow1, warp, B, C, H, H, -dt_s);
        const int ref_ch[5] = {2 * C + 2, 128, 64, 32, 2};
        // virtual concat [fe1 | warped fe2 | flow1] -> refine conv chain
        field_conv_host(fe1, warp, flow1, C, C, ref_ch, 4, refp[i],
                        flow1, fout, tmpA, tmpB, B, H, stream);
    }

    // ---- final full-resolution refinement ----
    resize_bilinear_kernel<<<g1(B * 2 * 256 * 256), 256, 0, stream>>>(
        out + oofs[3], xres, B, 2, 128, 128, 256, 256);
    const int up_ch[4] = {4, 64, 32, 2};
    // virtual concat [f1 | f2 | resized flow] -> up conv chain
    field_conv_host(f1, f2, xres, 1, 1, up_ch, 3, upp,
                    xres, out + oofs[4], tmpA, tmpB, B, 256, stream);

    (void)in_sizes; (void)out_size; (void)ws_size;
}
